// SlotAttentionPosEmbed_62199716380685
// MI455X (gfx1250) — matrix operand order repaired and checked by reading, not compiled, not run on hardware
//
#include <hip/hip_runtime.h>

// ---------------------------------------------------------------------------
// Types / helpers
// ---------------------------------------------------------------------------
typedef __attribute__((ext_vector_type(16))) __bf16 bf16x16;
typedef __attribute__((ext_vector_type(8)))  float  f32x8;

union FragAB {
    bf16x16 v;
    uint4   u4[2];
    unsigned short h[16];
};

__device__ __forceinline__ float bf2f(unsigned short h) {
    union { unsigned int u; float f; } c;
    c.u = ((unsigned int)h) << 16;
    return c.f;
}
__device__ __forceinline__ unsigned short f2bf(float f) {
    union { float f; unsigned int u; } c;
    c.f = f;
    unsigned int u = c.u;
    unsigned int r = u + 0x7FFFu + ((u >> 16) & 1u);   // round-to-nearest-even
    return (unsigned short)(r >> 16);
}

__device__ __forceinline__ f32x8 wmma_bf16(bf16x16 a, bf16x16 b, f32x8 c) {
    // emits v_wmma_f32_16x16x32_bf16
    return __builtin_amdgcn_wmma_f32_16x16x32_bf16(
        /*neg_a=*/false, a, /*neg_b=*/false, b,
        /*c_mod=*/(short)0, c, /*reuse_a=*/false, /*reuse_b=*/false);
}

__device__ __forceinline__ void lds_fence() {
    // CDNA5 split dependency counter: wait for all outstanding LDS ops
    asm volatile("s_wait_dscnt 0" ::: "memory");
}

// ---------------------------------------------------------------------------
// Problem constants
// ---------------------------------------------------------------------------
constexpr int kB  = 32;
constexpr int kH  = 128;
constexpr int kW  = 128;
constexpr int kHW = kH * kW;       // 16384
constexpr int kKC = 25 * 64;       // 1600, conv implicit-GEMM K
constexpr int kWST = 1608;         // padded LDS row stride (u16) for conv weights
constexpr int kQST = 72;           // padded LDS row stride (u16) for 64-wide mats

// ---------------------------------------------------------------------------
// Weight prep: conv w -> [O][tap*64+i] bf16 ; 64x64 mats -> [N][K] bf16
// ---------------------------------------------------------------------------
__global__ __launch_bounds__(256) void prep_kernel(
    const float* __restrict__ w2, const float* __restrict__ w3,
    const float* __restrict__ m1, const float* __restrict__ m2,
    const float* __restrict__ wk, const float* __restrict__ wv,
    unsigned short* __restrict__ Wt2, unsigned short* __restrict__ Wt3,
    unsigned short* __restrict__ Wm1, unsigned short* __restrict__ Wm2,
    unsigned short* __restrict__ Wkk, unsigned short* __restrict__ Wvv)
{
    int e = blockIdx.x * 256 + threadIdx.x;
    if (e < 64 * kKC) {
        int o = e / kKC, k = e - o * kKC;
        int tap = k >> 6, i = k & 63;
        Wt2[e] = f2bf(w2[(o * 64 + i) * 25 + tap]);
        Wt3[e] = f2bf(w3[(o * 64 + i) * 25 + tap]);
    }
    if (e < 64 * 64) {
        int n = e >> 6, k = e & 63;
        Wm1[e] = f2bf(m1[k * 64 + n]);
        Wm2[e] = f2bf(m2[k * 64 + n]);
        Wkk[e] = f2bf(wk[k * 64 + n]);
        Wvv[e] = f2bf(wv[k * 64 + n]);
    }
}

// ---------------------------------------------------------------------------
// conv1: 1 -> 64 channels, 5x5 SAME, relu; direct conv, out NHWC bf16
// ---------------------------------------------------------------------------
__global__ __launch_bounds__(256) void conv1_kernel(
    const float* __restrict__ data, const float* __restrict__ w1,
    const float* __restrict__ b1, unsigned short* __restrict__ act1)
{
    __shared__ float ws[64 * 25];
    for (int i = threadIdx.x; i < 64 * 25; i += 256) ws[i] = w1[i];
    __syncthreads();
    int p = blockIdx.x * 256 + threadIdx.x;     // pixel id, exact grid
    int b = p >> 14;
    int hw = p & (kHW - 1);
    int h = hw >> 7, w = hw & 127;
    float acc[64];
#pragma unroll
    for (int c = 0; c < 64; ++c) acc[c] = b1[c];
    for (int tap = 0; tap < 25; ++tap) {
        int ky = tap / 5, kx = tap - ky * 5;
        int hh = h + ky - 2, ww = w + kx - 2;
        if ((unsigned)hh < (unsigned)kH && (unsigned)ww < (unsigned)kW) {
            float dv = data[((size_t)b << 14) + (hh << 7) + ww];
#pragma unroll
            for (int c = 0; c < 64; ++c) acc[c] += dv * ws[c * 25 + tap];
        }
    }
    unsigned short* o = act1 + (size_t)p * 64;
#pragma unroll
    for (int c = 0; c < 64; ++c) o[c] = f2bf(fmaxf(acc[c], 0.0f));
}

// ---------------------------------------------------------------------------
// Implicit-GEMM conv (64->64, 5x5 SAME) via WMMA bf16.
//   - 32-pixel M tile per wave (2 A-frags, 8 accumulators): every LDS
//     B-fragment feeds two WMMAs, 8 independent accumulation chains.
//   - ky/kx loops with incremental addressing (no div/mod in steady state).
//   - wave-uniform interior fast path: no per-lane bounds exec masking.
//   - full 200 KB weight matrix staged in WGP LDS (320 KB available).
// ---------------------------------------------------------------------------
template<bool CHECK>
__device__ __forceinline__ void conv_accum_tile(
    const unsigned short* __restrict__ act,
    const unsigned short* __restrict__ wlm,   // wl + m*kWST
    int b, int h, int w0, int m, int hi, int kb0,
    f32x8 acc0[4], f32x8 acc1[4])
{
    const size_t rowStride = (size_t)kW * 64;
    // pointer to (hh = h-2, ww = w0+m-2) pixel, channel 0
    const unsigned short* rowp =
        act + ((size_t)((b * kH + (h - 2)) * kW) + (w0 + m - 2)) * 64;

    for (int ky = 0; ky < 5; ++ky) {
        int hh = h + ky - 2;
        bool rowok = !CHECK || ((unsigned)hh < (unsigned)kH);
        // warm cache for next row of this tile (global_prefetch path)
        __builtin_prefetch(rowp + rowStride, 0, 1);
        const unsigned short* p0 = rowp;
#pragma unroll
        for (int kx = 0; kx < 5; ++kx) {
            int ww0 = w0 + m + kx - 2;
            bool ok0 = !CHECK || (rowok && ((unsigned)ww0 < (unsigned)kW));
            bool ok1 = !CHECK || (rowok && ((unsigned)(ww0 + 16) < (unsigned)kW));
            const unsigned short* a0p = p0;
            const unsigned short* a1p = p0 + 16 * 64;
            int kwb = (ky * 5 + kx) * 64 + hi * 8;
#pragma unroll
            for (int half = 0; half < 2; ++half) {
                FragAB a0, a1;
                if (ok0) {
                    a0.u4[0] = *(const uint4*)(a0p + half * 32 + kb0);
                    a0.u4[1] = *(const uint4*)(a0p + half * 32 + kb0 + 16);
                } else {
                    a0.u4[0] = make_uint4(0u, 0u, 0u, 0u);
                    a0.u4[1] = make_uint4(0u, 0u, 0u, 0u);
                }
                if (ok1) {
                    a1.u4[0] = *(const uint4*)(a1p + half * 32 + kb0);
                    a1.u4[1] = *(const uint4*)(a1p + half * 32 + kb0 + 16);
                } else {
                    a1.u4[0] = make_uint4(0u, 0u, 0u, 0u);
                    a1.u4[1] = make_uint4(0u, 0u, 0u, 0u);
                }
                FragAB bb[4];
#pragma unroll
                for (int nt = 0; nt < 4; ++nt) {
                    const unsigned short* wb = wlm + nt * 16 * kWST + kwb + half * 32;
                    bb[nt].u4[0] = *(const uint4*)(wb);
                    bb[nt].u4[1] = *(const uint4*)(wb + 16);
                }
#pragma unroll
                for (int nt = 0; nt < 4; ++nt) {
                    acc0[nt] = wmma_bf16(a0.v, bb[nt].v, acc0[nt]);
                    acc1[nt] = wmma_bf16(a1.v, bb[nt].v, acc1[nt]);
                }
            }
            p0 += 64;           // next kx column
        }
        rowp += rowStride;      // next ky row
    }
}

__global__ __launch_bounds__(256) void conv_gemm_kernel(
    const unsigned short* __restrict__ act,
    const unsigned short* __restrict__ Wt,
    const float* __restrict__ bias,
    unsigned short* __restrict__ out_bf,
    float* __restrict__ out_f32,
    int relu)
{
    extern __shared__ unsigned short smem_cv[];
    unsigned short* wl = smem_cv;                       // [64][kWST]
    int tid = threadIdx.x;
    for (int i = tid; i < 64 * 200; i += 256) {         // 200 chunks of 8 bf16
        int o = i / 200, kc = (i - o * 200) * 8;
        *(uint4*)(wl + o * kWST + kc) = *(const uint4*)(Wt + o * kKC + kc);
    }
    __syncthreads();

    const int lane = tid & 31;
    const int m  = lane & 15;       // pixel-in-tile (A) / column (B,C)
    const int hi = lane >> 4;       // half-wave select
    const int kb0 = hi * 8;         // A-frag K base within 32-chunk
    const unsigned short* wlm = wl + m * kWST;
    const int gw = blockIdx.x * 8 + (tid >> 5);
    const int nw = gridDim.x * 8;
    const int TILES = kB * kH * (kW / 32);   // 32-pixel tiles

    for (int t = gw; t < TILES; t += nw) {
        int b = t / (kH * (kW / 32));
        int rem = t - b * (kH * (kW / 32));
        int h = rem / (kW / 32);
        int w0 = (rem - h * (kW / 32)) * 32;

        f32x8 acc0[4] = {}, acc1[4] = {};
        bool interior = (h >= 2) && (h <= kH - 3) && (w0 >= 2) && (w0 <= kW - 34);
        if (interior)
            conv_accum_tile<false>(act, wlm, b, h, w0, m, hi, kb0, acc0, acc1);
        else
            conv_accum_tile<true>(act, wlm, b, h, w0, m, hi, kb0, acc0, acc1);

        size_t pixbase = (size_t)(b * kH + h) * kW + w0;
#pragma unroll
        for (int nt = 0; nt < 4; ++nt) {
            int n = nt * 16 + m;
            float bn = bias[n];
#pragma unroll
            for (int j = 0; j < 8; ++j) {
                int mm = hi * 8 + j;
                float v0 = acc0[nt][j] + bn;
                float v1 = acc1[nt][j] + bn;
                if (relu) { v0 = fmaxf(v0, 0.0f); v1 = fmaxf(v1, 0.0f); }
                size_t i0 = (pixbase + mm) * 64 + n;
                size_t i1 = (pixbase + 16 + mm) * 64 + n;
                if (out_bf)  { out_bf[i0]  = f2bf(v0); out_bf[i1]  = f2bf(v1); }
                if (out_f32) { out_f32[i0] = v0;       out_f32[i1] = v1; }
            }
        }
    }
}

// ---------------------------------------------------------------------------
// Fused: pos-embed + LayerNorm + MLP(relu) + MLP + keys/vals projections.
// x3 f32 [B*HW,64] -> keys/vals bf16 [B*HW,64]. Per-wave LDS tile between
// WMMA GEMM stages, s_wait_dscnt fences for the LDS turnarounds.
// ---------------------------------------------------------------------------
__device__ __forceinline__ void load_a_frag(const unsigned short* tile,
                                            int m, int kb0, int c0, FragAB& a) {
    const unsigned short* p = tile + m * kQST + c0 + kb0;
    a.u4[0] = *(const uint4*)(p);
    a.u4[1] = *(const uint4*)(p + 16);
}
__device__ __forceinline__ void load_b_frag(const unsigned short* wl,
                                            int n, int hi, int c0, FragAB& b) {
    const unsigned short* p = wl + n * kQST + c0 + hi * 8;
    b.u4[0] = *(const uint4*)(p);
    b.u4[1] = *(const uint4*)(p + 16);
}
__device__ __forceinline__ void store_tile(unsigned short* tile, int m, int hi,
                                           const f32x8* acc, const float* bias,
                                           bool relu) {
#pragma unroll
    for (int nt = 0; nt < 4; ++nt) {
        int n = nt * 16 + m;
        float bn = bias[n];
#pragma unroll
        for (int j = 0; j < 8; ++j) {
            float x = acc[nt][j] + bn;
            if (relu) x = fmaxf(x, 0.0f);
            tile[(hi * 8 + j) * kQST + n] = f2bf(x);
        }
    }
}

__global__ __launch_bounds__(256) void fused_mlp_kv_kernel(
    const float* __restrict__ x3,
    const unsigned short* __restrict__ Wm1, const float* __restrict__ bm1,
    const unsigned short* __restrict__ Wm2, const float* __restrict__ bm2,
    const unsigned short* __restrict__ Wkk, const float* __restrict__ bkk,
    const unsigned short* __restrict__ Wvv, const float* __restrict__ bvv,
    const float* __restrict__ pos_w, const float* __restrict__ pos_b,
    const float* __restrict__ ln_g, const float* __restrict__ ln_b,
    unsigned short* __restrict__ keys, unsigned short* __restrict__ vals)
{
    extern __shared__ unsigned short smem_f[];
    unsigned short* w1l = smem_f;
    unsigned short* w2l = w1l + 64 * kQST;
    unsigned short* wkl = w2l + 64 * kQST;
    unsigned short* wvl = wkl + 64 * kQST;
    unsigned short* tileBase = wvl + 64 * kQST;

    int tid = threadIdx.x;
    for (int i = tid; i < 64 * 8; i += 256) {
        int n = i >> 3, kc = (i & 7) * 8;
        *(uint4*)(w1l + n * kQST + kc) = *(const uint4*)(Wm1 + n * 64 + kc);
        *(uint4*)(w2l + n * kQST + kc) = *(const uint4*)(Wm2 + n * 64 + kc);
        *(uint4*)(wkl + n * kQST + kc) = *(const uint4*)(Wkk + n * 64 + kc);
        *(uint4*)(wvl + n * kQST + kc) = *(const uint4*)(Wvv + n * 64 + kc);
    }
    __syncthreads();

    int lane = tid & 31;
    int m = lane & 15, hi = lane >> 4;
    int kb0 = hi * 8;
    unsigned short* tile = tileBase + (tid >> 5) * 16 * kQST;

    int gw = blockIdx.x * 8 + (tid >> 5);
    int nw = gridDim.x * 8;
    const int RT = (kB * kHW) / 16;

    for (int rt = gw; rt < RT; rt += nw) {
        // ---- positional embed + LayerNorm (2 lanes per row, shfl_xor 16)
        int row = rt * 16 + m;
        int hw = row & (kHW - 1);
        int ip = hw >> 7, jp = hw & 127;
        float gi = (float)ip * (1.0f / (kH - 1));
        float gj = (float)jp * (1.0f / (kW - 1));
        int c0h = hi * 32;
        const float* xr = x3 + (size_t)row * 64 + c0h;
        float v[32];
        float s = 0.0f, q = 0.0f;
#pragma unroll
        for (int i = 0; i < 32; ++i) {
            int c = c0h + i;
            float pv = gi * pos_w[c] + gj * pos_w[64 + c] +
                       (1.0f - gi) * pos_w[128 + c] + (1.0f - gj) * pos_w[192 + c] +
                       pos_b[c];
            float x = xr[i] + pv;
            v[i] = x; s += x; q += x * x;
        }
        s += __shfl_xor(s, 16, 32);
        q += __shfl_xor(q, 16, 32);
        float mu = s * (1.0f / 64.0f);
        float var = q * (1.0f / 64.0f) - mu * mu;
        float rs = rsqrtf(var + 1e-5f);
#pragma unroll
        for (int i = 0; i < 32; ++i) {
            int c = c0h + i;
            tile[m * kQST + c] = f2bf((v[i] - mu) * rs * ln_g[c] + ln_b[c]);
        }
        lds_fence();

        FragAB a0, a1;
        // ---- GEMM1: h1 = relu(xn @ W1 + b1)
        load_a_frag(tile, m, kb0, 0, a0);
        load_a_frag(tile, m, kb0, 32, a1);
        {
            f32x8 acc[4] = {};
#pragma unroll
            for (int nt = 0; nt < 4; ++nt) {
                FragAB b0, b1f;
                load_b_frag(w1l, nt * 16 + m, hi, 0, b0);
                load_b_frag(w1l, nt * 16 + m, hi, 32, b1f);
                acc[nt] = wmma_bf16(a0.v, b0.v, acc[nt]);
                acc[nt] = wmma_bf16(a1.v, b1f.v, acc[nt]);
            }
            lds_fence();
            store_tile(tile, m, hi, acc, bm1, true);
        }
        lds_fence();
        // ---- GEMM2: enc = h1 @ W2 + b2
        load_a_frag(tile, m, kb0, 0, a0);
        load_a_frag(tile, m, kb0, 32, a1);
        {
            f32x8 acc[4] = {};
#pragma unroll
            for (int nt = 0; nt < 4; ++nt) {
                FragAB b0, b1f;
                load_b_frag(w2l, nt * 16 + m, hi, 0, b0);
                load_b_frag(w2l, nt * 16 + m, hi, 32, b1f);
                acc[nt] = wmma_bf16(a0.v, b0.v, acc[nt]);
                acc[nt] = wmma_bf16(a1.v, b1f.v, acc[nt]);
            }
            lds_fence();
            store_tile(tile, m, hi, acc, bm2, false);
        }
        lds_fence();
        // ---- keys & vals projections (shared A-frags)
        load_a_frag(tile, m, kb0, 0, a0);
        load_a_frag(tile, m, kb0, 32, a1);
        f32x8 ak[4] = {}, av[4] = {};
#pragma unroll
        for (int nt = 0; nt < 4; ++nt) {
            FragAB bk0, bk1, bv0, bv1;
            load_b_frag(wkl, nt * 16 + m, hi, 0, bk0);
            load_b_frag(wkl, nt * 16 + m, hi, 32, bk1);
            load_b_frag(wvl, nt * 16 + m, hi, 0, bv0);
            load_b_frag(wvl, nt * 16 + m, hi, 32, bv1);
            ak[nt] = wmma_bf16(a0.v, bk0.v, ak[nt]);
            ak[nt] = wmma_bf16(a1.v, bk1.v, ak[nt]);
            av[nt] = wmma_bf16(a0.v, bv0.v, av[nt]);
            av[nt] = wmma_bf16(a1.v, bv1.v, av[nt]);
        }
#pragma unroll
        for (int nt = 0; nt < 4; ++nt) {
            int n = nt * 16 + m;
            float bkn = bkk[n], bvn = bvv[n];
#pragma unroll
            for (int j = 0; j < 8; ++j) {
                size_t r = (size_t)rt * 16 + hi * 8 + j;
                keys[r * 64 + n] = f2bf(ak[nt][j] + bkn);
                vals[r * 64 + n] = f2bf(av[nt][j] + bvn);
            }
        }
    }
}

// ---------------------------------------------------------------------------
// LayerNorm over queries [B*8, 64]
// ---------------------------------------------------------------------------
__global__ __launch_bounds__(256) void lnq_kernel(
    const float* __restrict__ queries, const float* __restrict__ g,
    const float* __restrict__ bb, float* __restrict__ qn)
{
    int r = blockIdx.x * 256 + threadIdx.x;
    if (r >= kB * 8) return;
    const float* x = queries + r * 64;
    float s = 0.0f, q = 0.0f;
    for (int i = 0; i < 64; ++i) { float v = x[i]; s += v; q += v * v; }
    float mu = s * (1.0f / 64.0f);
    float var = q * (1.0f / 64.0f) - mu * mu;
    float rs = rsqrtf(var + 1e-5f);
    for (int i = 0; i < 64; ++i)
        qn[r * 64 + i] = (x[i] - mu) * rs * g[i] + bb[i];
}

// ---------------------------------------------------------------------------
// Attention: logits[b,s,d] = qn[b,s,:].keys[b,d,:] * T, softmax over s,
// plus per-(b,s) pixel-sum reduction (LDS ds_add_f32 + global atomics)
// ---------------------------------------------------------------------------
__global__ __launch_bounds__(256) void attention_kernel(
    const float* __restrict__ qn, const unsigned short* __restrict__ keys,
    float* __restrict__ att, float* __restrict__ sums, float Tscale)
{
    __shared__ float qs[8 * 64];
    __shared__ float ssum[8];
    int b = blockIdx.x >> 6;                       // HW/256 = 64 blocks per batch
    int d = ((blockIdx.x & 63) << 8) + threadIdx.x;
    for (int i = threadIdx.x; i < 512; i += 256) qs[i] = qn[b * 512 + i];
    if (threadIdx.x < 8) ssum[threadIdx.x] = 0.0f;
    __syncthreads();

    const unsigned int* kr = (const unsigned int*)(keys + ((size_t)b * kHW + d) * 64);
    float lg[8] = {0, 0, 0, 0, 0, 0, 0, 0};
    for (int cc = 0; cc < 32; ++cc) {
        unsigned int u = kr[cc];
        float k0 = bf2f((unsigned short)u);
        float k1 = bf2f((unsigned short)(u >> 16));
#pragma unroll
        for (int s = 0; s < 8; ++s)
            lg[s] += k0 * qs[s * 64 + 2 * cc] + k1 * qs[s * 64 + 2 * cc + 1];
    }
    float mx = lg[0];
#pragma unroll
    for (int s = 1; s < 8; ++s) mx = fmaxf(mx, lg[s]);
    float e[8], se = 0.0f;
#pragma unroll
    for (int s = 0; s < 8; ++s) { e[s] = __expf((lg[s] - mx) * Tscale); se += e[s]; }
    float inv = 1.0f / se;
#pragma unroll
    for (int s = 0; s < 8; ++s) {
        float a = e[s] * inv;
        att[((size_t)(b * 8 + s)) * kHW + d] = a;
        atomicAdd(&ssum[s], a);
    }
    __syncthreads();
    if (threadIdx.x < 8) atomicAdd(&sums[b * 8 + threadIdx.x], ssum[threadIdx.x]);
}

// ---------------------------------------------------------------------------
// updates[b,s,c] = sum_d (att/sum + 1e-8) * vals[b,d,c]
// ---------------------------------------------------------------------------
__global__ __launch_bounds__(512) void updates_kernel(
    const float* __restrict__ att, const float* __restrict__ sums,
    const unsigned short* __restrict__ vals, float* __restrict__ upd)
{
    int b = blockIdx.x;
    int s = threadIdx.x >> 6;
    int c = threadIdx.x & 63;
    float inv = 1.0f / sums[b * 8 + s];
    const float* ar = att + (size_t)(b * 8 + s) * kHW;
    const unsigned short* vb = vals + (size_t)b * kHW * 64 + c;
    float acc = 0.0f;
    for (int d = 0; d < kHW; ++d)
        acc += (ar[d] * inv + 1e-8f) * bf2f(vb[(size_t)d * 64]);
    upd[(b * 8 + s) * 64 + c] = acc;
}

// ---------------------------------------------------------------------------
// GRUCell: one block per (b,s) row, 64 threads
// ---------------------------------------------------------------------------
__global__ __launch_bounds__(64) void gru_kernel(
    const float* __restrict__ upd, float* __restrict__ queries,
    const float* __restrict__ W_ih, const float* __restrict__ b_ih,
    const float* __restrict__ W_hh, const float* __restrict__ b_hh)
{
    __shared__ float xs[64], hs[64];
    int r = blockIdx.x;
    int j = threadIdx.x;
    xs[j] = upd[r * 64 + j];
    hs[j] = queries[r * 64 + j];
    __syncthreads();
    float ir = b_ih[j], iz = b_ih[64 + j], in_ = b_ih[128 + j];
    float hr = b_hh[j], hz = b_hh[64 + j], hn = b_hh[128 + j];
    for (int k = 0; k < 64; ++k) {
        float x = xs[k], h = hs[k];
        ir  += x * W_ih[j * 64 + k];
        iz  += x * W_ih[(64 + j) * 64 + k];
        in_ += x * W_ih[(128 + j) * 64 + k];
        hr  += h * W_hh[j * 64 + k];
        hz  += h * W_hh[(64 + j) * 64 + k];
        hn  += h * W_hh[(128 + j) * 64 + k];
    }
    float rr = 1.0f / (1.0f + __expf(-(ir + hr)));
    float zz = 1.0f / (1.0f + __expf(-(iz + hz)));
    float nn = tanhf(in_ + rr * hn);
    queries[r * 64 + j] = (1.0f - zz) * nn + zz * hs[j];
}

// ---------------------------------------------------------------------------
// wts = att / sums + 1e-8
// ---------------------------------------------------------------------------
__global__ __launch_bounds__(256) void wts_kernel(
    const float* __restrict__ att, const float* __restrict__ sums,
    float* __restrict__ wts)
{
    size_t i = (size_t)blockIdx.x * 256 + threadIdx.x;
    int rs = (int)(i >> 14);      // / HW
    wts[i] = att[i] / sums[rs] + 1e-8f;
}

__global__ __launch_bounds__(256) void copyf_kernel(const float* __restrict__ s,
                                                    float* __restrict__ d, int n) {
    int i = blockIdx.x * 256 + threadIdx.x;
    if (i < n) d[i] = s[i];
}
__global__ __launch_bounds__(256) void zerof_kernel(float* __restrict__ d, int n) {
    int i = blockIdx.x * 256 + threadIdx.x;
    if (i < n) d[i] = 0.0f;
}

// ---------------------------------------------------------------------------
// Host launcher
// ---------------------------------------------------------------------------
extern "C" void kernel_launch(void* const* d_in, const int* in_sizes, int n_in,
                              void* d_out, int out_size, void* d_ws, size_t ws_size,
                              hipStream_t stream)
{
    const float* data  = (const float*)d_in[0];
    const float* slots = (const float*)d_in[1];
    const float* cw1   = (const float*)d_in[2];
    const float* cb1   = (const float*)d_in[3];
    const float* cw2   = (const float*)d_in[4];
    const float* cb2   = (const float*)d_in[5];
    const float* cw3   = (const float*)d_in[6];
    const float* cb3   = (const float*)d_in[7];
    const float* posw  = (const float*)d_in[8];
    const float* posb  = (const float*)d_in[9];
    const float* dng   = (const float*)d_in[10];
    const float* dnb   = (const float*)d_in[11];
    const float* mw1   = (const float*)d_in[12];
    const float* mb1   = (const float*)d_in[13];
    const float* mw2   = (const float*)d_in[14];
    const float* mb2   = (const float*)d_in[15];
    const float* qg    = (const float*)d_in[16];
    const float* qb    = (const float*)d_in[17];
    const float* wk    = (const float*)d_in[18];
    const float* bk    = (const float*)d_in[19];
    const float* wv    = (const float*)d_in[20];
    const float* bv    = (const float*)d_in[21];
    const float* wih   = (const float*)d_in[22];
    const float* bih   = (const float*)d_in[23];
    const float* whh   = (const float*)d_in[24];
    const float* bhh   = (const float*)d_in[25];

    char* ws = (char*)d_ws;
    unsigned short* Wt2 = (unsigned short*)(ws + 0);        // 204800 B
    unsigned short* Wt3 = (unsigned short*)(ws + 204800);
    unsigned short* Wm1 = (unsigned short*)(ws + 409600);   // 8192 B each
    unsigned short* Wm2 = (unsigned short*)(ws + 417792);
    unsigned short* Wkk = (unsigned short*)(ws + 425984);
    unsigned short* Wvv = (unsigned short*)(ws + 434176);
    float* queries = (float*)(ws + 458752);                 // 64 KB
    float* qn      = (float*)(ws + 524288);                 // 64 KB
    float* upd     = (float*)(ws + 589824);                 // 64 KB
    float* sums    = (float*)(ws + 655360);                 // 1 KB
    float* att_t   = (float*)(ws + (1ull << 20));           // 16 MB
    unsigned short* act1 = (unsigned short*)(ws + (32ull << 20));   // 64 MB
    unsigned short* act2 = (unsigned short*)(ws + (96ull << 20));   // 64 MB
    float* x3            = (float*)(ws + (160ull << 20));           // 128 MB
    unsigned short* keysb = act1;   // reuse: act1 dead after conv2
    unsigned short* valsb = act2;   // reuse: act2 dead after conv3

    float* out     = (float*)d_out;
    float* out_q   = out;                        // [32*8*64]
    float* out_att = out + 16384;                // [32*8*16384]
    float* out_wts = out + 16384 + 4194304;      // [32*8*16384]

    const float Tscale = 1.25f;                  // 10/sqrt(64)

    prep_kernel<<<400, 256, 0, stream>>>(cw2, cw3, mw1, mw2, wk, wv,
                                         Wt2, Wt3, Wm1, Wm2, Wkk, Wvv);
    conv1_kernel<<<2048, 256, 0, stream>>>(data, cw1, cb1, act1);

    size_t csh = (size_t)64 * kWST * sizeof(unsigned short);    // ~201 KB LDS
    conv_gemm_kernel<<<512, 256, csh, stream>>>(act1, Wt2, cb2, act2, nullptr, 1);
    conv_gemm_kernel<<<512, 256, csh, stream>>>(act2, Wt3, cb3, nullptr, x3, 0);

    size_t fsh = (size_t)(4 * 64 * kQST + 8 * 16 * kQST) * sizeof(unsigned short);
    fused_mlp_kv_kernel<<<1024, 256, fsh, stream>>>(x3, Wm1, mb1, Wm2, mb2,
                                                    Wkk, bk, Wvv, bv,
                                                    posw, posb, dng, dnb,
                                                    keysb, valsb);

    copyf_kernel<<<64, 256, 0, stream>>>(slots, queries, 16384);
    for (int it = 0; it < 2; ++it) {
        lnq_kernel<<<1, 256, 0, stream>>>(queries, qg, qb, qn);
        zerof_kernel<<<1, 256, 0, stream>>>(sums, 256);
        attention_kernel<<<2048, 256, 0, stream>>>(qn, keysb, att_t, sums, Tscale);
        updates_kernel<<<32, 512, 0, stream>>>(att_t, sums, valsb, upd);
        gru_kernel<<<256, 64, 0, stream>>>(upd, queries, wih, bih, whh, bhh);
    }
    lnq_kernel<<<1, 256, 0, stream>>>(queries, qg, qb, qn);
    zerof_kernel<<<1, 256, 0, stream>>>(sums, 256);
    attention_kernel<<<2048, 256, 0, stream>>>(qn, keysb, out_att, sums, Tscale);
    wts_kernel<<<16384, 256, 0, stream>>>(out_att, sums, out_wts);
    copyf_kernel<<<64, 256, 0, stream>>>(queries, out_q, 16384);
}
